// MarginEmbeddingLoss_46566035423266
// MI455X (gfx1250) — matrix operand order repaired and checked by reading, not compiled, not run on hardware
//
#include <hip/hip_runtime.h>
#include <hip/hip_bf16.h>

typedef __attribute__((ext_vector_type(16))) _Float16 v16h;
typedef __attribute__((ext_vector_type(8)))  _Float16 v8h;
typedef __attribute__((ext_vector_type(8)))  float    v8f;

#define NPTS 8192
#define KDIM 256
#define MARGIN_F 1.0f

// ---------------------------------------------------------------------------
// Kernel 1: convert z (fp32) -> f16 workspace copy, compute row squared norms,
// and zero the global accumulator (block 0 / thread 0).
// One 256-thread block per row (KDIM == 256).
// ---------------------------------------------------------------------------
__global__ __launch_bounds__(256) void prep_kernel(
    const float* __restrict__ z,
    _Float16* __restrict__ z16,
    float* __restrict__ sqn,
    double* __restrict__ accum)
{
    const int row = blockIdx.x;
    const int tid = threadIdx.x;

    float v = z[row * KDIM + tid];
    z16[row * KDIM + tid] = (_Float16)v;

    float s = v * v;
    // wave32 reduction
    #pragma unroll
    for (int off = 16; off > 0; off >>= 1)
        s += __shfl_xor(s, off, 32);

    __shared__ float lds[8];
    const int wid  = tid >> 5;
    const int lane = tid & 31;
    if (lane == 0) lds[wid] = s;
    __syncthreads();
    if (tid == 0) {
        float t = 0.f;
        #pragma unroll
        for (int w = 0; w < 8; ++w) t += lds[w];
        sqn[row] = t;
        if (row == 0) *accum = 0.0;
    }
}

// ---------------------------------------------------------------------------
// Kernel 2: fused Gram-matrix WMMA GEMM + contrastive-loss epilogue.
// Block = 256 threads = 8 waves arranged 4 (M) x 2 (N); each wave computes a
// 32x32 output tile as 2x2 v_wmma_f32_16x16x32_f16 accumulators.
// Block tile: 128 (rows) x 64 (cols). Nothing is written except one f64
// atomic partial sum per block.
// ---------------------------------------------------------------------------
__device__ __forceinline__ float tile_epilogue(
    const v8f& c, int i0, int jn, float sqnj, int yj,
    const float* __restrict__ sqn, const int* __restrict__ y)
{
    float acc = 0.f;
    #pragma unroll
    for (int r = 0; r < 8; ++r) {
        const int i = i0 + r;            // M = r + 8*half folded into i0
        const float g  = c[r];           // <z_i, z_j>
        float sq = sqn[i] + sqnj - 2.0f * g;
        sq = fmaxf(sq, 0.0f);            // dist^2
        const bool same = (y[i] == yj);
        if (same) {
            if (i != jn) acc += sq;      // pos: dist^2 (diagonal excluded)
        } else {
            const float t = fmaxf(MARGIN_F - sqrtf(sq), 0.0f);
            acc += t * t;                // neg: relu(margin - dist)^2
        }
    }
    return acc;
}

__global__ __launch_bounds__(256) void gram_loss_kernel(
    const _Float16* __restrict__ z16,
    const float* __restrict__ sqn,
    const int* __restrict__ y,
    double* __restrict__ accum)
{
    const int lane = threadIdx.x & 31;
    const int wid  = threadIdx.x >> 5;
    const int half = lane >> 4;      // lane 16..31 selects second K/M half
    const int l16  = lane & 15;

    const int waveM = wid & 3;       // 4 wave rows
    const int waveN = wid >> 2;      // 2 wave cols
    const int rowBase = blockIdx.x * 128 + waveM * 32;
    const int colBase = blockIdx.y * 64  + waveN * 32;

    v8f c00 = {}, c01 = {}, c10 = {}, c11 = {};

    // A fragment rows: M = l16 (both halves); half selects K offset (+16)
    const _Float16* aptr0 = z16 + (size_t)(rowBase + l16)      * KDIM;
    const _Float16* aptr1 = z16 + (size_t)(rowBase + 16 + l16) * KDIM;
    // B fragment cols: N = l16; half selects K offset (+16)
    const _Float16* bptr0 = z16 + (size_t)(colBase + l16)      * KDIM;
    const _Float16* bptr1 = z16 + (size_t)(colBase + 16 + l16) * KDIM;

    for (int kk = 0; kk < KDIM; kk += 32) {
        // --- A fragments (16x32 f16 layout: elems 0-15 = K 16*half+0..15) ---
        v16h a0, a1;
        {
            v8h lo = *(const v8h*)(aptr0 + kk + 16 * half);
            v8h hi = *(const v8h*)(aptr0 + kk + 16 * half + 8);
            #pragma unroll
            for (int e = 0; e < 8; ++e) { a0[e] = lo[e]; a0[8 + e] = hi[e]; }
        }
        {
            v8h lo = *(const v8h*)(aptr1 + kk + 16 * half);
            v8h hi = *(const v8h*)(aptr1 + kk + 16 * half + 8);
            #pragma unroll
            for (int e = 0; e < 8; ++e) { a1[e] = lo[e]; a1[8 + e] = hi[e]; }
        }
        // --- B fragments (32x16 f16 layout: 16 contiguous K at 16*half) ---
        v16h b0 = *(const v16h*)(bptr0 + kk + 16 * half);
        v16h b1 = *(const v16h*)(bptr1 + kk + 16 * half);

        c00 = __builtin_amdgcn_wmma_f32_16x16x32_f16(false, a0, false, b0,
                                                     (short)0, c00, false, false);
        c01 = __builtin_amdgcn_wmma_f32_16x16x32_f16(false, a0, false, b1,
                                                     (short)0, c01, false, false);
        c10 = __builtin_amdgcn_wmma_f32_16x16x32_f16(false, a1, false, b0,
                                                     (short)0, c10, false, false);
        c11 = __builtin_amdgcn_wmma_f32_16x16x32_f16(false, a1, false, b1,
                                                     (short)0, c11, false, false);
    }

    // -------- epilogue: mask, relu, square, accumulate --------
    const int jn0   = colBase + l16;
    const int jn1   = colBase + 16 + l16;
    const float sj0 = sqn[jn0];
    const float sj1 = sqn[jn1];
    const int   yj0 = y[jn0];
    const int   yj1 = y[jn1];
    const int   i00 = rowBase + 8 * half;        // tile row 0
    const int   i10 = rowBase + 16 + 8 * half;   // tile row 1

    float local = 0.f;
    local += tile_epilogue(c00, i00, jn0, sj0, yj0, sqn, y);
    local += tile_epilogue(c01, i00, jn1, sj1, yj1, sqn, y);
    local += tile_epilogue(c10, i10, jn0, sj0, yj0, sqn, y);
    local += tile_epilogue(c11, i10, jn1, sj1, yj1, sqn, y);

    // wave32 reduce -> LDS float atomic -> one f64 global atomic per block
    #pragma unroll
    for (int off = 16; off > 0; off >>= 1)
        local += __shfl_xor(local, off, 32);

    __shared__ float bsum;
    if (threadIdx.x == 0) bsum = 0.f;
    __syncthreads();
    if (lane == 0) atomicAdd(&bsum, local);
    __syncthreads();
    if (threadIdx.x == 0) atomicAdd(accum, (double)bsum);
}

// ---------------------------------------------------------------------------
// Kernel 3: finalize. Denominator is exactly N^2 - N (+1e-8): the diagonal
// cancels between pos_mask and neg_mask sums.
// ---------------------------------------------------------------------------
__global__ void finalize_kernel(const double* __restrict__ accum,
                                float* __restrict__ out)
{
    const double denom = (double)NPTS * (double)(NPTS - 1) + 1e-8;
    out[0] = (float)(*accum / denom);
}

extern "C" void kernel_launch(void* const* d_in, const int* in_sizes, int n_in,
                              void* d_out, int out_size, void* d_ws, size_t ws_size,
                              hipStream_t stream) {
    (void)in_sizes; (void)n_in; (void)out_size; (void)ws_size;

    const float* z = (const float*)d_in[0];
    const int*   y = (const int*)d_in[1];

    // workspace layout: [0] f64 accum, [256] f32 sqn[N], [33024] f16 z16[N*K]
    char*     ws    = (char*)d_ws;
    double*   accum = (double*)ws;
    float*    sqn   = (float*)(ws + 256);
    _Float16* z16   = (_Float16*)(ws + 256 + NPTS * sizeof(float));

    prep_kernel<<<NPTS, 256, 0, stream>>>(z, z16, sqn, accum);

    dim3 grid(NPTS / 128, NPTS / 64);
    gram_loss_kernel<<<grid, 256, 0, stream>>>(z16, sqn, y, accum);

    finalize_kernel<<<1, 1, 0, stream>>>(accum, (float*)d_out);
}
